// SwinCrossScaleDecoder_18468359373137
// MI455X (gfx1250) — matrix operand order repaired and checked by reading, not compiled, not run on hardware
//
#include <hip/hip_runtime.h>
#include <math.h>

// ---------------------------------------------------------------------------
// CDNA5 (gfx1250) SwinCrossScaleDecoder — bf16 WMMA implementation, wave32.
// ---------------------------------------------------------------------------

typedef __attribute__((ext_vector_type(16))) __bf16 v16bf;
typedef __attribute__((ext_vector_type(8)))  __bf16 v8bf;
typedef __attribute__((ext_vector_type(8)))  float  v8f;
typedef __attribute__((ext_vector_type(4), aligned(4))) float f32x4u; // dword-aligned vec4

static __device__ __forceinline__ v16bf cat8(v8bf lo, v8bf hi) {
  return __builtin_shufflevector(lo, hi, 0,1,2,3,4,5,6,7,8,9,10,11,12,13,14,15);
}

// ---------------------------------------------------------------------------
// GEMM: Y[M,N] = act(X[M,K] @ W[K,N] + bias) (+ resid). W optionally [N,K]
// (wtrans=1). Block = 8 waves; each wave owns a 32(M)x64(N) strip = 2x4 WMMA
// tiles (8 v_wmma per staged panel); 32x64 bf16 B panel staged in LDS via two
// straight-line global_load_b128 per thread (fast path), reused by all 8
// waves.  All four B fragments are loaded into distinct registers BEFORE the
// WMMA burst so the ds_load_b128s drain with staggered s_wait_dscnt instead
// of full waits.  Out-of-range rows/cols are CLAMPED (results never stored),
// so the steady-state loop has no branches.
// ---------------------------------------------------------------------------
__global__ __launch_bounds__(256)
void wmma_gemm_kernel(const float* __restrict__ X, const float* __restrict__ W,
                      const float* __restrict__ bias, const float* __restrict__ resid,
                      float* __restrict__ Y, int M, int N, int K, int act, int wtrans)
{
  __shared__ __align__(16) __bf16 Bs[64][40]; // [n][k], 80B rows: conflict-free frag reads

  const int lane = threadIdx.x & 31;
  const int wave = threadIdx.x >> 5;
  const int n0   = blockIdx.x * 64;
  const int m0   = (blockIdx.y * 8 + wave) * 32;     // 2 M-tiles per wave
  const int hl   = lane >> 4;
  const int l15  = lane & 15;
  int ar0 = m0 + l15;      ar0 = (ar0 < M) ? ar0 : (M - 1);
  int ar1 = m0 + 16 + l15; ar1 = (ar1 < M) ? ar1 : (M - 1);
  const float* Xr0 = X + (size_t)ar0 * (size_t)K;
  const float* Xr1 = X + (size_t)ar1 * (size_t)K;
  const int kbA = hl ? 8 : 0;       // A lane K-base (ISA 7.12.2 16-bit A layout)
  const int kbB = hl ? 8 : 0;       // B lane K-base
  const int kfull = K & ~31;
  const bool nfull = (n0 + 63 < N); // block-uniform: whole panel in range

  v8f acc[2][4] = {{{}, {}, {}, {}}, {{}, {}, {}, {}}};

  int k0 = 0;
  for (; k0 < kfull; k0 += 32) {             // ---- unguarded main loop ----
    // ---- stage B panel (32 x 64) into LDS as bf16 ----
    if (!wtrans) {
      if (nfull) {                            // two b128 loads, straight-line
        const int kk = threadIdx.x >> 4;              // 0..15
        const int nn = (threadIdx.x & 15) * 4;        // 0..60
        const float* wp = W + (size_t)(k0 + kk) * N + (n0 + nn);
        f32x4u w0 = *(const f32x4u*)wp;
        f32x4u w1 = *(const f32x4u*)(wp + (size_t)16 * N);
#pragma unroll
        for (int e = 0; e < 4; ++e) {
          Bs[nn + e][kk]      = (__bf16)w0[e];
          Bs[nn + e][kk + 16] = (__bf16)w1[e];
        }
      } else {                                // boundary block: clamped scalar
        for (int t = threadIdx.x; t < 2048; t += 256) {
          int nn = t & 63, kk = t >> 6;
          int nc = n0 + nn; nc = (nc < N) ? nc : (N - 1);
          Bs[nn][kk] = (__bf16)W[(size_t)(k0 + kk) * N + nc];
        }
      }
    } else {                                  // W is [N,K]; vec4 along K
      const int nn = threadIdx.x >> 3;                // 0..31
      const int k4 = (threadIdx.x & 7) * 4;           // 0..28
      const float* wp = W + (size_t)(n0 + nn) * K + (k0 + k4);
      f32x4u w0 = *(const f32x4u*)wp;
      f32x4u w1 = *(const f32x4u*)(wp + (size_t)32 * K);
      __bf16* p0 = &Bs[nn][k4];
      __bf16* p1 = &Bs[nn + 32][k4];
#pragma unroll
      for (int e = 0; e < 4; ++e) { p0[e] = (__bf16)w0[e]; p1[e] = (__bf16)w1[e]; }
    }
    __syncthreads();

    f32x4u a00 = *(const f32x4u*)(Xr0 + k0 + kbA);
    f32x4u a01 = *(const f32x4u*)(Xr0 + k0 + kbA + 4);
    f32x4u a02 = *(const f32x4u*)(Xr0 + k0 + kbA + 16);
    f32x4u a03 = *(const f32x4u*)(Xr0 + k0 + kbA + 20);
    f32x4u a10 = *(const f32x4u*)(Xr1 + k0 + kbA);
    f32x4u a11 = *(const f32x4u*)(Xr1 + k0 + kbA + 4);
    f32x4u a12 = *(const f32x4u*)(Xr1 + k0 + kbA + 16);
    f32x4u a13 = *(const f32x4u*)(Xr1 + k0 + kbA + 20);
    if (k0 + 32 < K) {
      __builtin_prefetch(Xr0 + k0 + 32 + kbA, 0, 3);
      __builtin_prefetch(Xr1 + k0 + 32 + kbA, 0, 3);
    }
    v16bf a0, a1;
#pragma unroll
    for (int e = 0; e < 4; ++e) {
      a0[e] = (__bf16)a00[e]; a0[4 + e] = (__bf16)a01[e];
      a0[8 + e] = (__bf16)a02[e]; a0[12 + e] = (__bf16)a03[e];
      a1[e] = (__bf16)a10[e]; a1[4 + e] = (__bf16)a11[e];
      a1[8 + e] = (__bf16)a12[e]; a1[12 + e] = (__bf16)a13[e];
    }
    // load all 4 B fragments first (distinct regs -> staggered dscnt waits)
    v16bf bf[4];
#pragma unroll
    for (int nt = 0; nt < 4; ++nt) {
      const __bf16* bp = &Bs[nt * 16 + l15][kbB];
      bf[nt] = cat8(*(const v8bf*)bp, *(const v8bf*)(bp + 16));
    }
#pragma unroll
    for (int nt = 0; nt < 4; ++nt) {
      acc[0][nt] = __builtin_amdgcn_wmma_f32_16x16x32_bf16(false, a0, false, bf[nt],
                                                           (short)0, acc[0][nt], false, false);
      acc[1][nt] = __builtin_amdgcn_wmma_f32_16x16x32_bf16(false, a1, false, bf[nt],
                                                           (short)0, acc[1][nt], false, false);
    }
    __syncthreads();
  }

  if (k0 < K) {                              // ---- guarded K tail (once) ----
    if (!wtrans) {
      for (int t = threadIdx.x; t < 2048; t += 256) {
        int nn = t & 63, kk = t >> 6;
        int k = k0 + kk;
        int nc = n0 + nn; nc = (nc < N) ? nc : (N - 1);
        Bs[nn][kk] = (__bf16)((k < K) ? W[(size_t)k * N + nc] : 0.f);
      }
    } else {
      for (int t = threadIdx.x; t < 2048; t += 256) {
        int kk = t & 31, nn = t >> 5;
        int k = k0 + kk;
        int nc = n0 + nn; nc = (nc < N) ? nc : (N - 1);
        Bs[nn][kk] = (__bf16)((k < K) ? W[(size_t)nc * K + k] : 0.f);
      }
    }
    __syncthreads();
    v16bf a0, a1;
#pragma unroll
    for (int e = 0; e < 16; ++e) {
      int k = k0 + kbA + ((e < 8) ? e : (8 + e));
      a0[e] = (__bf16)((k < K) ? Xr0[k] : 0.f);
      a1[e] = (__bf16)((k < K) ? Xr1[k] : 0.f);
    }
    v16bf bf[4];
#pragma unroll
    for (int nt = 0; nt < 4; ++nt) {
      const __bf16* bp = &Bs[nt * 16 + l15][kbB];
      bf[nt] = cat8(*(const v8bf*)bp, *(const v8bf*)(bp + 16));
    }
#pragma unroll
    for (int nt = 0; nt < 4; ++nt) {
      acc[0][nt] = __builtin_amdgcn_wmma_f32_16x16x32_bf16(false, a0, false, bf[nt],
                                                           (short)0, acc[0][nt], false, false);
      acc[1][nt] = __builtin_amdgcn_wmma_f32_16x16x32_bf16(false, a1, false, bf[nt],
                                                           (short)0, acc[1][nt], false, false);
    }
  }

  // ---- epilogue (C/D layout: row = m0 + mt*16 + 8*hl + j, col = n0+nt*16+l15)
#pragma unroll
  for (int nt = 0; nt < 4; ++nt) {
    int col = n0 + nt * 16 + l15;
    if (col < N) {
      float bb = bias ? bias[col] : 0.f;
#pragma unroll
      for (int mt = 0; mt < 2; ++mt) {
#pragma unroll
        for (int j = 0; j < 8; ++j) {
          int r = m0 + mt * 16 + hl * 8 + j;
          if (r < M) {
            float v = acc[mt][nt][j] + bb;
            if (act == 1) { // tanh GELU (jax.nn.gelu default)
              float u = 0.7978845608028654f * (v + 0.044715f * v * v * v);
              v = 0.5f * v * (1.0f + tanhf(u));
            }
            if (resid) v += resid[(size_t)r * N + col];
            Y[(size_t)r * N + col] = v;
          }
        }
      }
    }
  }
}

// ---------------------------------------------------------------------------
// LayerNorm: one wave per token.
// ---------------------------------------------------------------------------
__global__ __launch_bounds__(256)
void layernorm_kernel(const float* __restrict__ X, const float* __restrict__ g,
                      const float* __restrict__ b, float* __restrict__ Y,
                      int T, int C)
{
  int t = blockIdx.x * (blockDim.x >> 5) + (threadIdx.x >> 5);
  int lane = threadIdx.x & 31;
  if (t >= T) return;
  const float* x = X + (size_t)t * C;
  float s = 0.f, s2 = 0.f;
  for (int c = lane; c < C; c += 32) { float v = x[c]; s += v; s2 += v * v; }
#pragma unroll
  for (int m = 16; m; m >>= 1) { s += __shfl_xor(s, m, 32); s2 += __shfl_xor(s2, m, 32); }
  float mean = s / (float)C;
  float inv  = rsqrtf(s2 / (float)C - mean * mean + 1e-5f);
  float* y = Y + (size_t)t * C;
  for (int c = lane; c < C; c += 32) y[c] = (x[c] - mean) * inv * g[c] + b[c];
}

// ---------------------------------------------------------------------------
// Windowed MHSA, one wave per (window, head).  S^T = K@Q^T (WMMA) so softmax
// lands in the VGPR dim + one shfl_xor(16); softmaxed S^T is *already* the A
// fragment for O = P@V.  Token/channel indices are clamped (masked keys get
// P=0, so clamped V rows contribute exactly zero).  ww hardcoded to 4.
// ---------------------------------------------------------------------------
__global__ __launch_bounds__(256)
void wmma_attn_kernel(const float* __restrict__ QKV, float* __restrict__ O,
                      int Bb, int Wh, int Ww, int C, int hd,
                      int wh, int sh, int sw)
{
  const int lane = threadIdx.x & 31;
  const int wave = threadIdx.x >> 5;
  const int gidx = blockIdx.x * (blockDim.x >> 5) + wave;
  const int nh = Wh / wh, nw = Ww >> 2;
  const int nwork = Bb * nh * nw * 3;
  if (gidx >= nwork) return;                 // wave-uniform exit
  const int head = gidx % 3;
  int win = gidx / 3;
  const int iw = win % nw; win /= nw;
  const int ih = win % nh; const int b = win / nh;
  const int ntok = wh << 2;
  const int L = Wh * Ww;
  const int hl = lane >> 4, l15 = lane & 15;

  auto tok2l = [&](int t) -> int {           // window token -> rolled flat index
    int gh = ih * wh + (t >> 2) + sh; if (gh >= Wh) gh -= Wh;
    int gw = (iw << 2) + (t & 3) + sw; if (gw >= Ww) gw -= Ww;
    return gh * Ww + gw;
  };

  const float* qkv_b = QKV + (size_t)b * (size_t)L * (size_t)(3 * C);
  const int lk = tok2l((l15 < ntok) ? l15 : (ntok - 1));   // clamped lane token
  const float* krow = qkv_b + (size_t)lk * (3 * C) + C + head * hd;
  const float* qrow = qkv_b + (size_t)lk * (3 * C) + head * hd;
  const int kbA = hl ? 8 : 0, kbB = hl ? 8 : 0;

  // ---- S^T = K @ Q^T  (head channels zero-padded to 32) ----
  v8f sacc = {};
  const int kfull = hd & ~31;
  int k0 = 0;
  for (; k0 < kfull; k0 += 32) {             // hd = 32/64/128: unguarded
    f32x4u a0 = *(const f32x4u*)(krow + k0 + kbA);
    f32x4u a1 = *(const f32x4u*)(krow + k0 + kbA + 4);
    f32x4u a2 = *(const f32x4u*)(krow + k0 + kbA + 16);
    f32x4u a3 = *(const f32x4u*)(krow + k0 + kbA + 20);
    f32x4u q0 = *(const f32x4u*)(qrow + k0 + kbB);
    f32x4u q1 = *(const f32x4u*)(qrow + k0 + kbB + 4);
    f32x4u q2 = *(const f32x4u*)(qrow + k0 + kbB + 16);
    f32x4u q3 = *(const f32x4u*)(qrow + k0 + kbB + 20);
    v16bf a, bq;
#pragma unroll
    for (int e = 0; e < 4; ++e) {
      a[e] = (__bf16)a0[e];  a[4 + e] = (__bf16)a1[e];
      a[8 + e] = (__bf16)a2[e]; a[12 + e] = (__bf16)a3[e];
      bq[e] = (__bf16)q0[e]; bq[4 + e] = (__bf16)q1[e];
      bq[8 + e] = (__bf16)q2[e]; bq[12 + e] = (__bf16)q3[e];
    }
    sacc = __builtin_amdgcn_wmma_f32_16x16x32_bf16(false, a, false, bq,
                                                   (short)0, sacc, false, false);
  }
  if (k0 < hd) {                             // hd = 15/24: one guarded chunk
    v16bf a, bq;
#pragma unroll
    for (int e = 0; e < 16; ++e) {
      int d = k0 + kbA + ((e < 8) ? e : (8 + e));
      a[e] = (__bf16)((d < hd) ? krow[d] : 0.f);
      int d2 = k0 + kbB + ((e < 8) ? e : (8 + e));
      bq[e] = (__bf16)((d2 < hd) ? qrow[d2] : 0.f);
    }
    sacc = __builtin_amdgcn_wmma_f32_16x16x32_bf16(false, a, false, bq,
                                                   (short)0, sacc, false, false);
  }

  // ---- softmax over keys (query = l15); padded keys masked ----
  const float scale = rsqrtf((float)hd);
  float mx = -1e30f;
#pragma unroll
  for (int j = 0; j < 8; ++j) {
    int key = hl * 8 + j;
    float v = (key < ntok) ? sacc[j] * scale : -1e30f;
    sacc[j] = v; mx = fmaxf(mx, v);
  }
  mx = fmaxf(mx, __shfl_xor(mx, 16, 32));
  float sum = 0.f;
#pragma unroll
  for (int j = 0; j < 8; ++j) { float e = __expf(sacc[j] - mx); sacc[j] = e; sum += e; }
  sum += __shfl_xor(sum, 16, 32);
  const float rs = 1.f / sum;
#pragma unroll
  for (int j = 0; j < 8; ++j) sacc[j] *= rs;

  // ---- O = P @ V (rolled V row indices precomputed, wave-uniform -> SALU)
  int lkv[16];
#pragma unroll
  for (int e = 0; e < 16; ++e) lkv[e] = tok2l((e < ntok) ? e : (ntok - 1));

  v16bf ap = {};
#pragma unroll
  for (int j = 0; j < 8; ++j) ap[j] = (__bf16)sacc[j];   // K=16..31 stay zero

  for (int d0 = 0; d0 < hd; d0 += 16) {
    int dc = d0 + l15; dc = (dc < hd) ? dc : (hd - 1);   // clamped channel
    v16bf bv = {};
    {                                        // element e<8 carries key 8*hl+e; e>=8 (K=16..31) zero
#pragma unroll
      for (int e = 0; e < 8; ++e)
        bv[e] = (__bf16)qkv_b[(size_t)lkv[hl * 8 + e] * (3 * C) + 2 * C + head * hd + dc];
    }
    v8f oacc = {};
    oacc = __builtin_amdgcn_wmma_f32_16x16x32_bf16(false, ap, false, bv,
                                                   (short)0, oacc, false, false);
#pragma unroll
    for (int j = 0; j < 8; ++j) {
      int q = hl * 8 + j, d = d0 + l15;
      if (q < ntok && d < hd)
        O[((size_t)b * L + tok2l(q)) * (size_t)C + head * hd + d] = oacc[j];
    }
  }
}

// ---------------------------------------------------------------------------
// VQ helpers
// ---------------------------------------------------------------------------
__global__ __launch_bounds__(256)
void cbnorm_kernel(const float* __restrict__ CB, float* __restrict__ N2, int Kc, int C)
{
  int k = blockIdx.x * (blockDim.x >> 5) + (threadIdx.x >> 5);
  int lane = threadIdx.x & 31;
  if (k >= Kc) return;
  const float* cb = CB + (size_t)k * C;
  float s = 0.f;
  for (int c = lane; c < C; c += 32) s += cb[c] * cb[c];
#pragma unroll
  for (int m = 16; m; m >>= 1) s += __shfl_xor(s, m, 32);
  if (lane == 0) N2[k] = s;
}

__global__ __launch_bounds__(256)
void vq_kernel(const float* __restrict__ Z, const float* __restrict__ S,
               const float* __restrict__ CB, const float* __restrict__ N2,
               float* __restrict__ Dec, float* __restrict__ cmout,
               float* __restrict__ cbout, int T, int C, int L, int accum)
{
  int t = blockIdx.x * (blockDim.x >> 5) + (threadIdx.x >> 5);
  int lane = threadIdx.x & 31;
  if (t >= T) return;
  float best = 3.4e38f; int bidx = 0;
  for (int k = lane; k < 256; k += 32) {
    float d = N2[k] - 2.f * S[(size_t)t * 256 + k];
    if (d < best) { best = d; bidx = k; }
  }
#pragma unroll
  for (int m = 16; m; m >>= 1) {
    float ob = __shfl_xor(best, m, 32);
    int   oi = __shfl_xor(bidx, m, 32);
    if (ob < best || (ob == best && oi < bidx)) { best = ob; bidx = oi; }
  }
  const float* cb = CB + (size_t)bidx * C;
  const float* z  = Z + (size_t)t * C;
  float* d = Dec + (size_t)t * C;
  float sq = 0.f;
  for (int c = lane; c < C; c += 32) {
    float zq = cb[c];
    float df = z[c] - zq;
    sq += df * df;
    d[c] = accum ? (d[c] + zq) : zq;
  }
#pragma unroll
  for (int m = 16; m; m >>= 1) sq += __shfl_xor(sq, m, 32);
  if (lane == 0) {
    float v = sq / ((float)L * (float)C);
    atomicAdd(&cmout[t / L], v);
    atomicAdd(&cbout[t / L], v);
  }
}

// ---------------------------------------------------------------------------
// elementwise / permutes
// ---------------------------------------------------------------------------
__global__ void sub_kernel(const float* A, const float* B, float* Y, size_t n)
{
  size_t i = (size_t)blockIdx.x * blockDim.x + threadIdx.x;
  if (i < n) Y[i] = A[i] - B[i];
}
__global__ void zero_kernel(float* p, size_t n)
{
  size_t i = (size_t)blockIdx.x * blockDim.x + threadIdx.x;
  if (i < n) p[i] = 0.f;
}
__global__ void expand_perm_kernel(const float* __restrict__ G, float* __restrict__ X,
                                   int Bb, int Wh, int Ww, int Cout)
{
  size_t n = (size_t)Bb * Wh * Ww * 2 * Cout;
  size_t i = (size_t)blockIdx.x * blockDim.x + threadIdx.x;
  if (i >= n) return;
  int c = (int)(i % Cout); size_t r = i / Cout;
  int s = (int)(r % 2); r /= 2;
  int w = (int)(r % Ww); r /= Ww;
  int h = (int)(r % Wh); int b = (int)(r / Wh);
  X[(((size_t)b * (2 * Wh) + 2 * h + s) * Ww + w) * Cout + c] = G[i];
}
__global__ void deembed_perm_kernel(const float* __restrict__ Xp, float* __restrict__ R,
                                    int Bb, int Fq, int Ww)
{
  size_t n = (size_t)Bb * 2 * (Fq * 3) * (Ww * 2);
  size_t i = (size_t)blockIdx.x * blockDim.x + threadIdx.x;
  if (i >= n) return;
  int wc = (int)(i % (Ww * 2)); size_t r = i / (Ww * 2);
  int fr = (int)(r % (Fq * 3)); r /= (Fq * 3);
  int d = (int)(r % 2); int b = (int)(r / 2);
  int f = fr / 3, ph = fr % 3;
  int w = wc / 2, pw = wc % 2;
  R[i] = Xp[(((size_t)b * Fq + f) * Ww + w) * 12 + (ph * 4 + pw * 2 + d)];
}

// ---------------------------------------------------------------------------
// Host orchestration
// ---------------------------------------------------------------------------
extern "C" void kernel_launch(void* const* d_in, const int* in_sizes, int n_in,
                              void* d_out, int out_size, void* d_ws, size_t ws_size,
                              hipStream_t stream)
{
  (void)in_sizes; (void)n_in; (void)out_size; (void)ws_size;
  const int Bb = 16, Ww = 600;
  const int HD[5] = {384, 192, 96, 72, 45};
  const int OD[5] = {192, 96, 72, 45, 45};

  const float* enc[6]; for (int i = 0; i < 6; ++i) enc[i] = (const float*)d_in[i];
  const float* cbs[6]; for (int i = 0; i < 6; ++i) cbs[i] = (const float*)d_in[6 + i];
  auto P = [&](int idx) { return (const float*)d_in[idx]; };
  // blocks: base 12, 26 arrays/stage (b0:12, b1:12, exp_w, exp_b); post at 142;
  // de_w=166, de_b=167.  swin order: ln1_g,ln1_b,qkv_w,qkv_b,proj_w,proj_b,
  // ln2_g,ln2_b,w1,b1,w2,b2.

  float* ws = (float*)d_ws;
  size_t o = 0;
  float* DEC = ws + o; o += 28000000ull;
  float* LNB = ws + o; o += 28000000ull;
  float* QKV = ws + o; o += 83000000ull;
  float* HID = ws + o; o += 111000000ull;
  float* OAT = ws + o; o += 28000000ull;
  float* ZRS = ws + o; o += 7500000ull;
  float* SCR = ws + o; o += 10000000ull;
  float* CBN = ws + o; o += 256ull;

  float* recon = (float*)d_out;
  float* cmv = recon + 7372800;            // [16] cm, [16] cb, [16] kl
  zero_kernel<<<1, 64, 0, stream>>>(cmv, 48);

  auto gemm = [&](const float* X, const float* Wt, const float* bias,
                  const float* resid, float* Y, int M, int N, int K,
                  int act, int wtrans) {
    dim3 g((unsigned)((N + 63) / 64), (unsigned)((M + 255) / 256));
    wmma_gemm_kernel<<<g, 256, 0, stream>>>(X, Wt, bias, resid, Y, M, N, K, act, wtrans);
  };

  auto run_vq = [&](const float* Z, const float* CBp, int C, int T, int L, int accum) {
    cbnorm_kernel<<<32, 256, 0, stream>>>(CBp, CBN, 256, C);
    gemm(Z, CBp, nullptr, nullptr, SCR, T, 256, C, 0, 1);
    vq_kernel<<<(T + 7) / 8, 256, 0, stream>>>(Z, SCR, CBp, CBN, DEC,
                                               cmv, cmv + 16, T, C, L, accum);
  };

  auto run_block = [&](int C, int Wh, int shift, int pb) {
    const int L = Wh * Ww, T = Bb * L;
    layernorm_kernel<<<(T + 7) / 8, 256, 0, stream>>>(DEC, P(pb + 0), P(pb + 1), LNB, T, C);
    gemm(LNB, P(pb + 2), P(pb + 3), nullptr, QKV, T, 3 * C, C, 0, 0);
    const int wh = (Wh < 4) ? Wh : 4;
    const int sh = shift ? wh / 2 : 0, sw = shift ? 2 : 0;
    const int hd = C / 3;
    const int nwork = Bb * (Wh / wh) * (Ww / 4) * 3;
    wmma_attn_kernel<<<(nwork + 7) / 8, 256, 0, stream>>>(QKV, OAT, Bb, Wh, Ww, C,
                                                          hd, wh, sh, sw);
    gemm(OAT, P(pb + 4), P(pb + 5), DEC, DEC, T, C, C, 0, 0);
    layernorm_kernel<<<(T + 7) / 8, 256, 0, stream>>>(DEC, P(pb + 6), P(pb + 7), LNB, T, C);
    gemm(LNB, P(pb + 8), P(pb + 9), nullptr, HID, T, 4 * C, C, 1, 0);
    gemm(HID, P(pb + 10), P(pb + 11), DEC, DEC, T, C, 4 * C, 0, 0);
  };

  int Wh = 2;
  run_vq(enc[5], cbs[0], 384, Bb * Wh * Ww, Wh * Ww, 0);

  for (int i = 0; i < 5; ++i) {
    const int C = HD[i], L = Wh * Ww, T = Bb * L;
    const size_t nel = (size_t)T * C;
    sub_kernel<<<(unsigned)((nel + 255) / 256), 256, 0, stream>>>(enc[5 - i], DEC, ZRS, nel);
    run_vq(ZRS, cbs[i + 1], C, T, L, 1);
    const int pb = 12 + 26 * i;
    run_block(C, Wh, 0, pb);
    run_block(C, Wh, 1, pb + 12);
    const int Cout = OD[i];
    gemm(DEC, P(pb + 24), P(pb + 25), nullptr, HID, T, 2 * Cout, C, 0, 0);
    const size_t ne2 = (size_t)T * 2 * Cout;
    expand_perm_kernel<<<(unsigned)((ne2 + 255) / 256), 256, 0, stream>>>(HID, DEC, Bb, Wh, Ww, Cout);
    Wh *= 2;
  }

  run_block(45, Wh, 0, 142);
  run_block(45, Wh, 1, 154);

  const int T = Bb * Wh * Ww;
  gemm(DEC, P(166), P(167), nullptr, HID, T, 12, 45, 0, 0);
  const size_t nr = (size_t)Bb * 2 * 192 * (Ww * 2);
  deembed_perm_kernel<<<(unsigned)((nr + 255) / 256), 256, 0, stream>>>(HID, recon, Bb, 64, Ww);
}
